// MoELayer_13589276524736
// MI455X (gfx1250) — matrix operand order repaired and checked by reading, not compiled, not run on hardware
//
#include <hip/hip_runtime.h>

// ---------------- problem constants ----------------
#define N_TOK  8192      // B*S tokens
#define DMODEL 2048
#define DHALF  1024
#define DFF    8192
#define NEXP   8
#define TOPK   2
#define CHUNK  2048      // rows of one expert processed per (FC1,FC2) pair

// ---------------- WMMA / async vector types ----------------
typedef __attribute__((ext_vector_type(16))) __bf16 v16bf;
typedef __attribute__((ext_vector_type(8)))  float  v8f;
typedef __attribute__((ext_vector_type(4)))  int    v4i;

typedef __attribute__((address_space(1))) v4i* v4i_g;
typedef __attribute__((address_space(3))) v4i* v4i_l;

union FragAB { uint4 q[2]; v16bf m; };

// ---------------- helpers ----------------
__device__ __forceinline__ unsigned short f2bf(float f) {
    union { float f; unsigned u; } v; v.f = f;
    unsigned r = v.u + 0x7FFFu + ((v.u >> 16) & 1u);   // RNE
    return (unsigned short)(r >> 16);
}
__device__ __forceinline__ float bf2f(unsigned short s) {
    union { unsigned u; float f; } v; v.u = ((unsigned)s) << 16;
    return v.f;
}
__device__ __forceinline__ float gelu_exact(float x) {
    return 0.5f * x * (1.0f + erff(x * 0.70710678118654752f));
}

// ---- CDNA5 async global->LDS copy (ASYNCcnt-tracked), 16B per lane ----
__device__ __forceinline__ void async_copy_b128(const unsigned short* gptr,
                                                unsigned short* lptr /*shared*/)
{
#if __has_builtin(__builtin_amdgcn_global_load_async_to_lds_b128)
    __builtin_amdgcn_global_load_async_to_lds_b128(
        (v4i_g)(v4i*)(size_t)gptr,       // global src (const stripped via int cast)
        (v4i_l)(v4i*)lptr,               // LDS dst
        0, 0);
#else
    unsigned lds_off =
        (unsigned)(unsigned long long)(__attribute__((address_space(3))) void*)lptr;
    asm volatile("global_load_async_to_lds_b128 %0, %1, off"
                 :: "v"(lds_off), "v"(gptr) : "memory");
#endif
}
__device__ __forceinline__ void wait_async0()
{
#if __has_builtin(__builtin_amdgcn_s_wait_asynccnt)
    __builtin_amdgcn_s_wait_asynccnt(0);
#else
    asm volatile("s_wait_asynccnt 0" ::: "memory");
#endif
}

// =====================================================================
// Generic 128x128x32 bf16-WMMA GEMM, 256 threads = 8 wave32.
// A (bf16) tiles: async global->LDS (ASYNCcnt path).
// B (f32) tiles: VGPR-staged with on-the-fly bf16 convert, transposed in LDS.
// MODE 0: router hidden  H = gelu(Xbf @ Wr1 + br1)       -> bf16
// MODE 1: expert FC1     Hb = gelu(gather(Xbf) @ W1e+b1e) -> bf16
// MODE 2: expert FC2     out[token] += comb * (Hb @ W2e + b2e)
// =====================================================================
template<int MODE>
__global__ __launch_bounds__(256)
void gemm_kernel(const unsigned short* __restrict__ A,   // bf16, row-major, ld=lda
                 const float*          __restrict__ Bw,  // f32 [Kd, Nn] row-major
                 const float*          __restrict__ bias,// [Nn]
                 unsigned short*       __restrict__ outBf,
                 float*                __restrict__ outF,
                 const int*            __restrict__ idxListE,
                 const float*          __restrict__ combE, // comb + e, stride 8
                 const int*            __restrict__ cntE,
                 int chunkBase, int Kd, int Nn, int lda)
{
    __shared__ __align__(16) unsigned short As[2][128 * 40]; // 32 cols + 8 pad
    __shared__ __align__(16) unsigned short Bs[2][128 * 40]; // [N][K] transposed

    const int tid = threadIdx.x;
    const int bm = blockIdx.x, bn = blockIdx.y;

    int cnt = 0, activeRows = 1 << 30;
    if constexpr (MODE != 0) {
        cnt = *cntE;
        int rem = cnt - chunkBase;
        activeRows = rem < CHUNK ? rem : CHUNK;
        if (activeRows <= bm * 128) return;          // block-uniform early exit
    }

    // ----- A-tile staging: thread -> (row, 16-col half) -----
    const int rowA = tid >> 1, half = tid & 1;
    int aRow;
    if constexpr (MODE == 0) {
        aRow = bm * 128 + rowA;
    } else if constexpr (MODE == 1) {
        int lp = chunkBase + bm * 128 + rowA;
        aRow = (lp < cnt) ? idxListE[lp] : 0;        // gather token row
    } else {
        int r = bm * 128 + rowA;
        aRow = (r < activeRows) ? r : 0;             // chunk-local Hb row
    }
    const unsigned short* Ath = A + (size_t)aRow * lda + half * 16;

    // ----- B-tile staging: 4 slots of 4 consecutive f32 -----
    int kB[4], gB[4]; const float* Bth[4];
    #pragma unroll
    for (int s = 0; s < 4; ++s) {
        int slot = tid + 256 * s;
        kB[s] = slot >> 5; gB[s] = slot & 31;
        Bth[s] = Bw + (size_t)kB[s] * Nn + bn * 128 + gB[s] * 4;
    }

    v8f acc[2][4];
    #pragma unroll
    for (int mi = 0; mi < 2; ++mi)
        #pragma unroll
        for (int ni = 0; ni < 4; ++ni)
            #pragma unroll
            for (int j = 0; j < 8; ++j) acc[mi][ni][j] = 0.0f;

    const int KT = Kd / 32;
    float4 bq[4];

    auto asyncA = [&](int kt, int b) {
        const unsigned short* g = Ath + kt * 32;
        unsigned short* l = &As[b][rowA * 40 + half * 16];
        async_copy_b128(g,     l);
        async_copy_b128(g + 8, l + 8);
    };
    auto loadB = [&](int kt) {
        #pragma unroll
        for (int s = 0; s < 4; ++s) {
            const float* gp = Bth[s] + (size_t)kt * 32 * Nn;
            bq[s] = *(const float4*)gp;
            __builtin_prefetch(gp + (size_t)32 * Nn, 0, 0); // speculative, next k-tile
        }
    };
    auto storeB = [&](int b) {
        #pragma unroll
        for (int s = 0; s < 4; ++s) {
            Bs[b][(gB[s] * 4 + 0) * 40 + kB[s]] = f2bf(bq[s].x);
            Bs[b][(gB[s] * 4 + 1) * 40 + kB[s]] = f2bf(bq[s].y);
            Bs[b][(gB[s] * 4 + 2) * 40 + kB[s]] = f2bf(bq[s].z);
            Bs[b][(gB[s] * 4 + 3) * 40 + kB[s]] = f2bf(bq[s].w);
        }
    };

    // wave/lane geometry (ISA 16-bit A / B VGPR layouts)
    const int lane = tid & 31, wv = tid >> 5;
    const int wm = wv >> 1, wn = wv & 1;         // 4x2 wave grid
    const int r  = lane & 15, hi = lane >> 4;

    // prologue: tile 0
    asyncA(0, 0);
    loadB(0); storeB(0);
    wait_async0();
    __syncthreads();

    int buf = 0;
    for (int kt = 0; kt < KT; ++kt) {
        if (kt + 1 < KT) { asyncA(kt + 1, buf ^ 1); loadB(kt + 1); }
        FragAB af[2], bfr[4];
        #pragma unroll
        for (int mi = 0; mi < 2; ++mi) {
            int base = (wm * 32 + mi * 16 + r) * 40 + hi * 8;   // K: hi*8..+7, 16+hi*8..+7
            af[mi].q[0] = *(const uint4*)&As[buf][base];
            af[mi].q[1] = *(const uint4*)&As[buf][base + 16];
        }
        #pragma unroll
        for (int ni = 0; ni < 4; ++ni) {
            int base = (wn * 64 + ni * 16 + r) * 40 + hi * 8;   // K: hi*8..+7, 16+hi*8..+7
            bfr[ni].q[0] = *(const uint4*)&Bs[buf][base];
            bfr[ni].q[1] = *(const uint4*)&Bs[buf][base + 16];
        }
        #pragma unroll
        for (int mi = 0; mi < 2; ++mi)
            #pragma unroll
            for (int ni = 0; ni < 4; ++ni)
                acc[mi][ni] = __builtin_amdgcn_wmma_f32_16x16x32_bf16(
                    false, af[mi].m, false, bfr[ni].m, (short)0, acc[mi][ni],
                    false, false);
        if (kt + 1 < KT) storeB(buf ^ 1);
        wait_async0();            // A tile for buf^1 has landed in LDS
        __syncthreads();
        buf ^= 1;
    }

    // ----- epilogue (C layout: vgpr i -> M = i + 8*hi, N = r) -----
    #pragma unroll
    for (int mi = 0; mi < 2; ++mi) {
        #pragma unroll
        for (int ni = 0; ni < 4; ++ni) {
            int col = bn * 128 + wn * 64 + ni * 16 + r;
            float bcol = bias[col];
            #pragma unroll
            for (int i = 0; i < 8; ++i) {
                int rowL = wm * 32 + mi * 16 + i + 8 * hi;
                float v = acc[mi][ni][i] + bcol;
                if constexpr (MODE == 0) {
                    int gR = bm * 128 + rowL;
                    outBf[(size_t)gR * Nn + col] = f2bf(gelu_exact(v));
                } else if constexpr (MODE == 1) {
                    int lp = chunkBase + bm * 128 + rowL;
                    if (lp < cnt)
                        outBf[(size_t)(bm * 128 + rowL) * Nn + col] = f2bf(gelu_exact(v));
                } else {
                    int rc = bm * 128 + rowL;
                    if (rc < activeRows) {
                        int t = idxListE[chunkBase + rc];
                        float cw = combE[t * NEXP];
                        outF[(size_t)t * Nn + col] += v * cw;   // experts run sequentially
                    }
                }
            }
        }
    }
}

// ---------------- small kernels ----------------
__global__ void init_kernel(float* __restrict__ out, int* __restrict__ cnt, size_t nTot)
{
    size_t i = (size_t)blockIdx.x * blockDim.x + threadIdx.x;
    size_t stride = (size_t)gridDim.x * blockDim.x;
    for (; i < nTot; i += stride) out[i] = 0.0f;
    size_t g = (size_t)blockIdx.x * blockDim.x + threadIdx.x;
    if (g < NEXP) cnt[g] = 0;
}

__global__ void cast_x_kernel(const float* __restrict__ x,
                              unsigned short* __restrict__ xbf, int n4)
{
    int i = blockIdx.x * blockDim.x + threadIdx.x;
    if (i >= n4) return;
    float4 v = ((const float4*)x)[i];
    uint2 o;
    o.x = (unsigned)f2bf(v.x) | ((unsigned)f2bf(v.y) << 16);
    o.y = (unsigned)f2bf(v.z) | ((unsigned)f2bf(v.w) << 16);
    ((uint2*)xbf)[i] = o;
}

// logits = Hr(bf16) @ Wr2 + br2 ; tiny N=8, plain VALU dot products
__global__ void logits_kernel(const unsigned short* __restrict__ Hr,
                              const float* __restrict__ Wr2,
                              const float* __restrict__ br2,
                              float* __restrict__ logits)
{
    int gid = blockIdx.x * blockDim.x + threadIdx.x;   // n*E threads
    int t = gid >> 3, e = gid & 7;
    const unsigned short* h = Hr + (size_t)t * DHALF;
    float s = br2[e];
    for (int i = 0; i < DHALF; ++i)
        s += bf2f(h[i]) * Wr2[(size_t)i * NEXP + e];
    logits[gid] = s;
}

// per-token top-2 softmax + dispatch lists + deterministic usage partials
__global__ void router_topk_kernel(const float* __restrict__ logits,
                                   float* __restrict__ comb,
                                   int* __restrict__ idxList,
                                   int* __restrict__ cnt,
                                   float* __restrict__ usagePart)
{
    __shared__ float sp[256 * NEXP];
    int t = blockIdx.x * 256 + threadIdx.x;            // n = 32*256 tokens
    float l[NEXP];
    #pragma unroll
    for (int e = 0; e < NEXP; ++e) l[e] = logits[t * NEXP + e];

    // full softmax (for load-balance usage)
    float m = l[0];
    #pragma unroll
    for (int e = 1; e < NEXP; ++e) m = fmaxf(m, l[e]);
    float p[NEXP], s = 0.0f;
    #pragma unroll
    for (int e = 0; e < NEXP; ++e) { p[e] = expf(l[e] - m); s += p[e]; }
    float inv = 1.0f / s;
    #pragma unroll
    for (int e = 0; e < NEXP; ++e) p[e] *= inv;

    // top-2 (ties -> lowest index, matches lax.top_k)
    int i0 = 0;
    #pragma unroll
    for (int e = 1; e < NEXP; ++e) if (l[e] > l[i0]) i0 = e;
    int i1 = (i0 == 0) ? 1 : 0;
    #pragma unroll
    for (int e = 0; e < NEXP; ++e) if (e != i0 && l[e] > l[i1]) i1 = e;

    float d  = expf(l[i1] - l[i0]);
    float w1 = d / (1.0f + d), w0 = 1.0f / (1.0f + d);

    #pragma unroll
    for (int e = 0; e < NEXP; ++e) comb[t * NEXP + e] = 0.0f;
    comb[t * NEXP + i0] = w0;
    comb[t * NEXP + i1] = w1;

    int p0 = atomicAdd(&cnt[i0], 1); idxList[i0 * N_TOK + p0] = t;
    int p1 = atomicAdd(&cnt[i1], 1); idxList[i1 * N_TOK + p1] = t;

    // fixed-order block reduction of usage (deterministic)
    #pragma unroll
    for (int e = 0; e < NEXP; ++e) sp[threadIdx.x * NEXP + e] = p[e];
    __syncthreads();
    if (threadIdx.x < NEXP) {
        float acc = 0.0f;
        for (int j = 0; j < 256; ++j) acc += sp[j * NEXP + threadIdx.x];
        usagePart[blockIdx.x * NEXP + threadIdx.x] = acc;
    }
}

__global__ void lb_loss_kernel(const float* __restrict__ usagePart,
                               float* __restrict__ outScalar, int nBlocks)
{
    if (threadIdx.x != 0 || blockIdx.x != 0) return;
    float u[NEXP];
    for (int e = 0; e < NEXP; ++e) {
        float s = 0.0f;
        for (int b = 0; b < nBlocks; ++b) s += usagePart[b * NEXP + e];
        u[e] = s / (float)N_TOK;
    }
    float mean = 0.0f;
    for (int e = 0; e < NEXP; ++e) mean += u[e];
    mean /= (float)NEXP;
    float var = 0.0f;
    for (int e = 0; e < NEXP; ++e) { float dd = u[e] - mean; var += dd * dd; }
    var /= (float)(NEXP - 1);                 // unbiased, like torch.var
    outScalar[0] = 0.01f * var;
}

// ---------------- host orchestration ----------------
extern "C" void kernel_launch(void* const* d_in, const int* in_sizes, int n_in,
                              void* d_out, int out_size, void* d_ws, size_t ws_size,
                              hipStream_t stream)
{
    const float* x   = (const float*)d_in[0];
    const float* Wr1 = (const float*)d_in[1];
    const float* br1 = (const float*)d_in[2];
    const float* Wr2 = (const float*)d_in[3];
    const float* br2 = (const float*)d_in[4];
    const float* W1  = (const float*)d_in[5];
    const float* b1  = (const float*)d_in[6];
    const float* W2  = (const float*)d_in[7];
    const float* b2  = (const float*)d_in[8];
    float* out = (float*)d_out;

    // workspace layout (all offsets 16B aligned)
    char* ws = (char*)d_ws;
    size_t off = 0;
    unsigned short* Xbf  = (unsigned short*)(ws + off); off += (size_t)N_TOK * DMODEL * 2; // 33.5MB
    unsigned short* Hr   = (unsigned short*)(ws + off); off += (size_t)N_TOK * DHALF * 2;  // 16.8MB
    unsigned short* Hbuf = (unsigned short*)(ws + off); off += (size_t)CHUNK * DFF * 2;    // 33.5MB
    float* logits        = (float*)(ws + off);          off += (size_t)N_TOK * NEXP * 4;
    float* comb          = (float*)(ws + off);          off += (size_t)N_TOK * NEXP * 4;
    int*   idxList       = (int*)(ws + off);            off += (size_t)NEXP * N_TOK * 4;
    int*   cnt           = (int*)(ws + off);            off += 256;
    float* usagePart     = (float*)(ws + off);          off += (size_t)32 * NEXP * 4;

    const size_t nD = (size_t)N_TOK * DMODEL;

    // 1) zero output + counters
    init_kernel<<<8192, 256, 0, stream>>>(out, cnt, nD);
    // 2) x -> bf16
    cast_x_kernel<<<(int)(nD / 4 / 256), 256, 0, stream>>>(x, Xbf, (int)(nD / 4));
    // 3) router hidden: [8192,2048] x [2048,1024], gelu -> Hr
    gemm_kernel<0><<<dim3(N_TOK / 128, DHALF / 128), 256, 0, stream>>>(
        Xbf, Wr1, br1, Hr, nullptr, nullptr, nullptr, nullptr, 0, DMODEL, DHALF, DMODEL);
    // 4) logits
    logits_kernel<<<(N_TOK * NEXP) / 256, 256, 0, stream>>>(Hr, Wr2, br2, logits);
    // 5) top-2 / combine / dispatch / usage partials
    router_topk_kernel<<<N_TOK / 256, 256, 0, stream>>>(logits, comb, idxList, cnt, usagePart);
    // 6) load-balance loss -> out[nD]
    lb_loss_kernel<<<1, 32, 0, stream>>>(usagePart, out + nD, N_TOK / 256);

    // 7) experts, sequential (safe non-atomic accumulation into out)
    for (int e = 0; e < NEXP; ++e) {
        const float* W1e = W1 + (size_t)e * DMODEL * DFF;
        const float* b1e = b1 + (size_t)e * DFF;
        const float* W2e = W2 + (size_t)e * DFF * DMODEL;
        const float* b2e = b2 + (size_t)e * DMODEL;
        const int*   idxE = idxList + (size_t)e * N_TOK;
        const int*   cntE = cnt + e;
        const float* combE = comb + e;
        for (int c = 0; c < N_TOK / CHUNK; ++c) {
            int chunkBase = c * CHUNK;
            // FC1: gather(Xbf) [<=2048,2048] x W1e [2048,8192], gelu -> Hbuf
            gemm_kernel<1><<<dim3(CHUNK / 128, DFF / 128), 256, 0, stream>>>(
                Xbf, W1e, b1e, Hbuf, nullptr, idxE, nullptr, cntE,
                chunkBase, DMODEL, DFF, DMODEL);
            // FC2: Hbuf [<=2048,8192] x W2e [8192,2048] -> scaled scatter-add
            gemm_kernel<2><<<dim3(CHUNK / 128, DMODEL / 128), 256, 0, stream>>>(
                Hbuf, W2e, b2e, nullptr, out, idxE, combE, cntE,
                chunkBase, DFF, DMODEL, DFF);
        }
    }
    (void)in_sizes; (void)n_in; (void)out_size; (void)ws_size;
}
